// SelfAttention_81587198755253
// MI455X (gfx1250) — matrix operand order repaired and checked by reading, not compiled, not run on hardware
//
#include <hip/hip_runtime.h>

// Self-attention fwd: x (4096, 4, 1024) fp32 -> out (4096, 4, 1024) fp32
// Flash-attention tiling on CDNA5 WMMA (bf16 operands, fp32 accumulate).

#define SEQ     4096
#define NBAT    4
#define DIM     1024
#define ROWSTR  (NBAT * DIM)      // element stride between seq rows
#define BR      64                // query rows per workgroup
#define BC      32                // key/value rows per iteration
#define NIT     (SEQ / BC)        // 128 iterations
#define THREADS 512               // 16 wave32
#define CROWS   1032              // krow padded row stride (bf16 elems, 2064 B)
#define KTROW   40                // kt padded row stride (bf16 elems, 80 B, 16B-aligned)
#define SCALE   0.03125f          // 1/sqrt(1024)

typedef __attribute__((ext_vector_type(16))) __bf16    v16bf;
typedef __attribute__((ext_vector_type(2)))  __bf16    bf16x2;
typedef __attribute__((ext_vector_type(8)))  float     v8f;
typedef __attribute__((ext_vector_type(4)))  float     f32x4;
typedef __attribute__((ext_vector_type(4)))  unsigned  u32x4;
typedef __attribute__((ext_vector_type(2)))  unsigned  u32x2;

union ABits {
    v16bf          v;
    u32x4          q[2];
    unsigned       u[8];
    unsigned short h[16];
};

struct __align__(16) SMem {
    unsigned short krow[BC][CROWS];   // K tile, row-major bf16 (padded vs bank conflicts)
    unsigned short kt[DIM][KTROW];    // K tile, transposed bf16 (padded, 16B-aligned rows)
    float          s[BR][BC];         // reduced raw scores for this iteration
    unsigned short p[BR][BC];         // softmax probabilities, bf16
    float          m[BR];             // running row max
    float          l[BR];             // running row sum
    float          scl[BR];           // accumulator rescale factor this iter
};

// fp32 pair -> packed bf16 pair; __bf16 casts should lower to v_cvt_pk_bf16_f32
static __device__ __forceinline__ unsigned pk2bf(float a, float b) {
    bf16x2 t;
    t[0] = (__bf16)a;
    t[1] = (__bf16)b;
    return __builtin_bit_cast(unsigned, t);
}
static __device__ __forceinline__ unsigned short f2bf(float f) {
    return (unsigned short)(pk2bf(f, 0.0f) & 0xffffu);
}
static __device__ __forceinline__ v8f zero8() {
    v8f z;
    #pragma unroll
    for (int i = 0; i < 8; ++i) z[i] = 0.0f;
    return z;
}

__global__ __launch_bounds__(THREADS, 1)
void attn_fwd_81587198755253(const float* __restrict__ x, float* __restrict__ out) {
    __shared__ SMem sm;

    const int tid  = threadIdx.x;
    const int lane = tid & 31;
    const int wave = tid >> 5;
    const int r    = wave >> 2;       // 0..3 : which 16-row slice of the Q block
    const int c    = wave & 3;        // 0..3 : which 256-wide dim chunk
    const int lm   = lane & 15;
    const int hi   = lane >> 4;       // lane half (selects K-slice per WMMA layout)
    const int b    = blockIdx.y;
    const int q0   = blockIdx.x * BR;

    // ---- Q block (16 rows x 256 dims per wave) -> bf16 A operands in registers ----
    ABits aq[8];
    {
        const float* qp = x + (size_t)(q0 + r * 16 + lm) * ROWSTR + (size_t)b * DIM;
        #pragma unroll
        for (int kc = 0; kc < 8; ++kc) {
            const int d0 = c * 256 + kc * 32 + hi * 16;
            f32x4 fa = *(const f32x4*)(qp + d0);
            f32x4 fb = *(const f32x4*)(qp + d0 + 4);
            f32x4 fc = *(const f32x4*)(qp + d0 + 8);
            f32x4 fd = *(const f32x4*)(qp + d0 + 12);
            aq[kc].u[0] = pk2bf(fa.x, fa.y);  aq[kc].u[1] = pk2bf(fa.z, fa.w);
            aq[kc].u[2] = pk2bf(fb.x, fb.y);  aq[kc].u[3] = pk2bf(fb.z, fb.w);
            aq[kc].u[4] = pk2bf(fc.x, fc.y);  aq[kc].u[5] = pk2bf(fc.z, fc.w);
            aq[kc].u[6] = pk2bf(fd.x, fd.y);  aq[kc].u[7] = pk2bf(fd.z, fd.w);
        }
    }

    v8f acc[16];                       // O accumulator: 16 rows x 256 dims per wave
    #pragma unroll
    for (int i = 0; i < 16; ++i) acc[i] = zero8();

    if (tid < BR) { sm.m[tid] = -3.0e38f; sm.l[tid] = 0.0f; }

    const int ldrow = tid >> 4;        // 0..31  K row this thread loads
    const int ldsub = tid & 15;        // 0..15  64-float span within the row

    // hoisted LDS bases (per-kc / per-sub deltas become immediate DS offsets)
    const unsigned short* k0b  = &sm.krow[lm     ][c * 256 + hi * 16];
    const unsigned short* k1b  = &sm.krow[lm + 16][c * 256 + hi * 16];
    const unsigned short* ktb  = &sm.kt[c * 256 + lm][hi * 16];

    for (int j = 0; j < NIT; ++j) {
        // ---- zero score buffer + cooperative K/V tile load (fp32 -> bf16, 2 layouts) ----
        {
            f32x4 z4 = {0.0f, 0.0f, 0.0f, 0.0f};
            ((f32x4*)&sm.s[0][0])[tid] = z4;                    // 512 * 16B = 8 KB

            const float* src = x + (size_t)(j * BC + ldrow) * ROWSTR + (size_t)b * DIM;
            #pragma unroll
            for (int i = 0; i < 16; ++i) {
                const int d = ldsub * 64 + i * 4;
                f32x4 f = *(const f32x4*)(src + d);
                unsigned u0 = pk2bf(f.x, f.y), u1 = pk2bf(f.z, f.w);
                u32x2 pr = {u0, u1};
                *(u32x2*)&sm.krow[ldrow][d] = pr;              // row-major copy
                sm.kt[d + 0][ldrow] = (unsigned short)(u0);    // transposed copy
                sm.kt[d + 1][ldrow] = (unsigned short)(u0 >> 16);
                sm.kt[d + 2][ldrow] = (unsigned short)(u1);
                sm.kt[d + 3][ldrow] = (unsigned short)(u1 >> 16);
            }
            if (j + 1 < NIT) __builtin_prefetch(src + (size_t)BC * ROWSTR, 0, 3);
        }
        __syncthreads();

        // ---- phase 1: S partials (split-d over c-waves), 16 WMMAs/wave, pipelined ----
        {
            v8f s0 = zero8(), s1 = zero8();
            ABits b0[2], b1[2];
            b0[0].q[0] = *(const u32x4*)(k0b);  b0[0].q[1] = *(const u32x4*)(k0b + 8);
            b1[0].q[0] = *(const u32x4*)(k1b);  b1[0].q[1] = *(const u32x4*)(k1b + 8);
            #pragma unroll
            for (int kc = 0; kc < 8; ++kc) {
                const int cur = kc & 1, nxt = cur ^ 1;
                if (kc + 1 < 8) {                               // prefetch next K-chunk
                    const unsigned short* n0 = k0b + (kc + 1) * 32;
                    const unsigned short* n1 = k1b + (kc + 1) * 32;
                    b0[nxt].q[0] = *(const u32x4*)(n0);  b0[nxt].q[1] = *(const u32x4*)(n0 + 8);
                    b1[nxt].q[0] = *(const u32x4*)(n1);  b1[nxt].q[1] = *(const u32x4*)(n1 + 8);
                }
                s0 = __builtin_amdgcn_wmma_f32_16x16x32_bf16(false, aq[kc].v, false, b0[cur].v,
                                                             (short)0, s0, false, false);
                s1 = __builtin_amdgcn_wmma_f32_16x16x32_bf16(false, aq[kc].v, false, b1[cur].v,
                                                             (short)0, s1, false, false);
            }
            #pragma unroll
            for (int v = 0; v < 8; ++v) {
                const int row = r * 16 + v + hi * 8;           // C-layout: M = v + 8*hi
                atomicAdd(&sm.s[row][lm     ], s0[v]);         // ds_add_f32 reduction
                atomicAdd(&sm.s[row][lm + 16], s1[v]);
            }
        }
        __syncthreads();

        // ---- phase 2: online softmax over the 32 new columns (thread per row) ----
        if (tid < BR) {
            const int row = tid;
            float mo = sm.m[row], lo = sm.l[row];
            float sb[BC], mx = -3.0e38f;
            #pragma unroll
            for (int t = 0; t < BC; ++t) { sb[t] = sm.s[row][t] * SCALE; mx = fmaxf(mx, sb[t]); }
            const float mn = fmaxf(mo, mx);
            const float sc = __expf(mo - mn);
            float sum = 0.0f;
            #pragma unroll
            for (int t = 0; t < BC; ++t) {
                const float pv = __expf(sb[t] - mn);
                sum += pv;
                sm.p[row][t] = f2bf(pv);
            }
            sm.m[row]   = mn;
            sm.l[row]   = lo * sc + sum;
            sm.scl[row] = sc;
        }
        __syncthreads();

        // ---- phase 3: rescale accumulators, O += P * V (16 WMMAs/wave, pipelined) ----
        {
            // load P A-operand first so it is in flight during the rescale multiplies
            const unsigned short* pp = &sm.p[r * 16 + lm][hi * 16];
            ABits pa;
            pa.q[0] = *(const u32x4*)(pp);                     // t = hi*16 .. +8
            pa.q[1] = *(const u32x4*)(pp + 8);                 // t = hi*16+8 .. +8

            float fac[8];
            #pragma unroll
            for (int v = 0; v < 8; ++v) fac[v] = sm.scl[r * 16 + v + hi * 8];
            #pragma unroll
            for (int i = 0; i < 16; ++i)
                #pragma unroll
                for (int v = 0; v < 8; ++v) acc[i][v] = acc[i][v] * fac[v];

            ABits bv[2];
            bv[0].q[0] = *(const u32x4*)(ktb);  bv[0].q[1] = *(const u32x4*)(ktb + 8);
            #pragma unroll
            for (int sub = 0; sub < 16; ++sub) {
                const int cur = sub & 1, nxt = cur ^ 1;
                if (sub + 1 < 16) {                             // prefetch next V subtile
                    const unsigned short* vp = ktb + (sub + 1) * (16 * KTROW);
                    bv[nxt].q[0] = *(const u32x4*)(vp);
                    bv[nxt].q[1] = *(const u32x4*)(vp + 8);
                }
                acc[sub] = __builtin_amdgcn_wmma_f32_16x16x32_bf16(false, pa.v, false, bv[cur].v,
                                                                   (short)0, acc[sub], false, false);
            }
        }
        __syncthreads();
    }

    // ---- epilogue: normalize by running sum l, store fp32 ----
    {
        float invl[8];
        #pragma unroll
        for (int v = 0; v < 8; ++v) invl[v] = 1.0f / sm.l[r * 16 + v + hi * 8];
        #pragma unroll
        for (int sub = 0; sub < 16; ++sub) {
            const int dc = c * 256 + sub * 16 + lm;
            #pragma unroll
            for (int v = 0; v < 8; ++v) {
                const size_t srow = (size_t)(q0 + r * 16 + v + hi * 8);
                out[srow * ROWSTR + (size_t)b * DIM + dc] = acc[sub][v] * invl[v];
            }
        }
    }
}

extern "C" void kernel_launch(void* const* d_in, const int* in_sizes, int n_in,
                              void* d_out, int out_size, void* d_ws, size_t ws_size,
                              hipStream_t stream) {
    (void)in_sizes; (void)n_in; (void)out_size; (void)d_ws; (void)ws_size;
    const float* x = (const float*)d_in[0];
    float* out = (float*)d_out;
    dim3 grid(SEQ / BR, NBAT, 1);
    dim3 block(THREADS, 1, 1);
    hipLaunchKernelGGL(attn_fwd_81587198755253, grid, block, 0, stream, x, out);
}
